// PairedStructureEncoder_87084756894163
// MI455X (gfx1250) — matrix operand order repaired and checked by reading, not compiled, not run on hardware
//
#include <hip/hip_runtime.h>
#include <stdint.h>

#define B_    4
#define VH_   130
#define VL_   120
#define EP_   256
#define L_    506
#define NTOK  2024   // B_ * L_
#define NPAD  2048   // padded token rows (multiple of 64)
#define DIN_  256
#define DEP_  128
#define D_    512
#define H_    16
#define NL_   8
#define HID_  2048

typedef __attribute__((ext_vector_type(16))) __bf16 v16bf;
typedef __attribute__((ext_vector_type(8)))  float  v8f;

static __device__ __forceinline__ uint16_t f2bf(float f) {
  uint32_t u = __builtin_bit_cast(uint32_t, f);
  u += 0x7FFFu + ((u >> 16) & 1u);      // round-to-nearest-even
  return (uint16_t)(u >> 16);
}

struct BfPair { uint4 lo; uint4 hi; };
static __device__ __forceinline__ v16bf ld16(const uint16_t* p0, const uint16_t* p1) {
  BfPair pr;
  pr.lo = *reinterpret_cast<const uint4*>(p0);
  pr.hi = *reinterpret_cast<const uint4*>(p1);
  return __builtin_bit_cast(v16bf, pr);
}

static __device__ __forceinline__ float softplusf(float x) {
  return logf(1.0f + __expf(x));
}

// ---------------------------------------------------------------------------
// fp32 -> bf16 conversion with optional row/col zero padding.
// src: (batch, sr, sc) row-major fp32; dst: (batch, dr, dc) row-major bf16.
// ---------------------------------------------------------------------------
__global__ void cvt_pad_kernel(const float* __restrict__ src, uint16_t* __restrict__ dst,
                               int batch, int sr, int dr, int sc, int dc) {
  long idx = (long)blockIdx.x * 256 + threadIdx.x;
  long total = (long)batch * dr * dc;
  if (idx >= total) return;
  int c = (int)(idx % dc);
  long t = idx / dc;
  int r = (int)(t % dr);
  int bb = (int)(t / dr);
  float v = 0.0f;
  if (r < sr && c < sc) v = src[((long)bb * sr + r) * sc + c];
  dst[idx] = f2bf(v);
}

__global__ void zero_f32_kernel(float* p, long n) {
  long i = (long)blockIdx.x * 256 + threadIdx.x;
  if (i < n) p[i] = 0.0f;
}
__global__ void zero_b16_kernel(uint16_t* p, long n) {
  long i = (long)blockIdx.x * 256 + threadIdx.x;
  if (i < n) p[i] = 0;
}

// ---------------------------------------------------------------------------
// Gather per-token rotation matrices, translations, masks into flat buffers.
// ---------------------------------------------------------------------------
__global__ void meta_kernel(const float* __restrict__ vh_rots, const float* __restrict__ vh_trans,
                            const float* __restrict__ vl_rots, const float* __restrict__ vl_trans,
                            const float* __restrict__ ep_rots, const float* __restrict__ ep_trans,
                            const unsigned char* __restrict__ vh_mask,
                            const unsigned char* __restrict__ vl_mask,
                            const unsigned char* __restrict__ ep_mask,
                            float* __restrict__ rotb, float* __restrict__ trb,
                            unsigned char* __restrict__ maskb) {
  int t = blockIdx.x * 256 + threadIdx.x;
  if (t >= NPAD) return;
  if (t >= NTOK) {
    maskb[t] = 1;
    for (int k = 0; k < 9; ++k) rotb[(long)t * 9 + k] = (k % 4 == 0) ? 1.0f : 0.0f;
    for (int k = 0; k < 3; ++k) trb[(long)t * 3 + k] = 0.0f;
    return;
  }
  int b = t / L_, l = t % L_;
  const float* R; const float* T; unsigned char m;
  if (l < VH_) {
    int p = l;
    R = vh_rots + ((long)b * VH_ + p) * 9; T = vh_trans + ((long)b * VH_ + p) * 3;
    m = vh_mask[b * VH_ + p];
  } else if (l < VH_ + VL_) {
    int p = l - VH_;
    R = vl_rots + ((long)b * VL_ + p) * 9; T = vl_trans + ((long)b * VL_ + p) * 3;
    m = vl_mask[b * VL_ + p];
  } else {
    int p = l - VH_ - VL_;
    R = ep_rots + ((long)b * EP_ + p) * 9; T = ep_trans + ((long)b * EP_ + p) * 3;
    m = ep_mask[b * EP_ + p];
  }
  for (int k = 0; k < 9; ++k) rotb[(long)t * 9 + k] = R[k];
  for (int k = 0; k < 3; ++k) trb[(long)t * 3 + k] = T[k];
  maskb[t] = m ? 1 : 0;
}

// ---------------------------------------------------------------------------
// Embedding: emb[t,:] = W @ x[t] (+ sinusoidal PE for vh/vl segments)
// ---------------------------------------------------------------------------
__global__ __launch_bounds__(256)
void embed_kernel(const float* __restrict__ vh_x, const float* __restrict__ vl_x,
                  const float* __restrict__ ep_x,
                  const float* __restrict__ Wvh, const float* __restrict__ Wvl,
                  const float* __restrict__ Wep, float* __restrict__ emb) {
  int t = blockIdx.x;              // 0..NTOK-1
  int b = t / L_, l = t % L_;
  const float* x; const float* W; int Kd; int p = -1;
  if (l < VH_)            { p = l;          x = vh_x + ((long)b * VH_ + p) * DIN_; W = Wvh; Kd = DIN_; }
  else if (l < VH_ + VL_) { p = l - VH_;    x = vl_x + ((long)b * VL_ + p) * DIN_; W = Wvl; Kd = DIN_; }
  else                    { int q = l - VH_ - VL_; x = ep_x + ((long)b * EP_ + q) * DEP_; W = Wep; Kd = DEP_; }
  __shared__ float xs[DIN_];
  if ((int)threadIdx.x < Kd) xs[threadIdx.x] = x[threadIdx.x];
  __syncthreads();
  for (int d = threadIdx.x; d < D_; d += 256) {
    const float* wr = W + (long)d * Kd;
    float acc = 0.0f;
    for (int k = 0; k < Kd; ++k) acc += wr[k] * xs[k];
    if (p >= 0) {
      float ang = (float)p * __powf(1000.0f, -2.0f * (float)d / 511.0f);
      acc += (d & 1) ? __cosf(ang) : __sinf(ang);
    }
    emb[(long)t * D_ + d] = acc;
  }
}

// ---------------------------------------------------------------------------
// LayerNorm -> bf16 (one block per row, 512 columns)
// ---------------------------------------------------------------------------
__global__ __launch_bounds__(256)
void ln_kernel(const float* __restrict__ x, const float* __restrict__ g,
               const float* __restrict__ bta, uint16_t* __restrict__ out) {
  int row = blockIdx.x, tid = threadIdx.x;
  const float* xr = x + (long)row * D_;
  float a0 = xr[tid], a1 = xr[tid + 256];
  __shared__ float red[256];
  red[tid] = a0 + a1;
  __syncthreads();
  for (int off = 128; off > 0; off >>= 1) {
    if (tid < off) red[tid] += red[tid + off];
    __syncthreads();
  }
  float mean = red[0] * (1.0f / 512.0f);
  __syncthreads();
  float d0 = a0 - mean, d1 = a1 - mean;
  red[tid] = d0 * d0 + d1 * d1;
  __syncthreads();
  for (int off = 128; off > 0; off >>= 1) {
    if (tid < off) red[tid] += red[tid + off];
    __syncthreads();
  }
  float inv = rsqrtf(red[0] * (1.0f / 512.0f) + 1e-5f);
  out[(long)row * D_ + tid]       = f2bf(d0 * inv * g[tid] + bta[tid]);
  out[(long)row * D_ + tid + 256] = f2bf(d1 * inv * g[tid + 256] + bta[tid + 256]);
}

// ---------------------------------------------------------------------------
// WMMA GEMM: C(M x N) = A(M x K, bf16) * Bw(N x K, bf16)^T
//   block = 4 waves; wave w -> rows [by*64+16w, +16), cols [bx*64, +64)
//   mode 0: C = acc ; mode 1: C += acc + bias[n] ; mode 2: C += acc
// M, N multiples of 64; K multiple of 32. A/B fragments use contiguous k = 16*half + s.
// ---------------------------------------------------------------------------
__global__ __launch_bounds__(128)
void gemm_wmma_kernel(const uint16_t* __restrict__ A, int lda,
                      const uint16_t* __restrict__ Bw, int ldb,
                      float* __restrict__ C, int ldc,
                      const float* __restrict__ bias, int K, int mode) {
  const int lane = threadIdx.x & 31;
  const int wave = threadIdx.x >> 5;
  const int half = lane >> 4;      // K-half selector
  const int l15  = lane & 15;
  const int m0 = blockIdx.y * 64 + wave * 16;
  const int n0 = blockIdx.x * 64;

  const uint16_t* arow = A + (long)(m0 + l15) * lda;
  const uint16_t* b0p = Bw + (long)(n0 + l15) * ldb + half * 16;
  const uint16_t* b1p = b0p + 16L * ldb;
  const uint16_t* b2p = b0p + 32L * ldb;
  const uint16_t* b3p = b0p + 48L * ldb;

  v8f acc0 = {}, acc1 = {}, acc2 = {}, acc3 = {};
  for (int k0 = 0; k0 < K; k0 += 32) {
    v16bf a  = ld16(arow + k0 + half * 16, arow + k0 + half * 16 + 8);
    v16bf b0 = ld16(b0p + k0, b0p + k0 + 8);
    v16bf b1 = ld16(b1p + k0, b1p + k0 + 8);
    v16bf b2 = ld16(b2p + k0, b2p + k0 + 8);
    v16bf b3 = ld16(b3p + k0, b3p + k0 + 8);
    acc0 = __builtin_amdgcn_wmma_f32_16x16x32_bf16(false, a, false, b0, (short)0, acc0, false, false);
    acc1 = __builtin_amdgcn_wmma_f32_16x16x32_bf16(false, a, false, b1, (short)0, acc1, false, false);
    acc2 = __builtin_amdgcn_wmma_f32_16x16x32_bf16(false, a, false, b2, (short)0, acc2, false, false);
    acc3 = __builtin_amdgcn_wmma_f32_16x16x32_bf16(false, a, false, b3, (short)0, acc3, false, false);
  }

  float bs0 = 0.0f, bs1 = 0.0f, bs2 = 0.0f, bs3 = 0.0f;
  if (mode == 1) {
    bs0 = bias[n0 + l15];      bs1 = bias[n0 + l15 + 16];
    bs2 = bias[n0 + l15 + 32]; bs3 = bias[n0 + l15 + 48];
  }
  const int crow0 = m0 + half * 8;
  for (int r = 0; r < 8; ++r) {
    float* cp = C + (long)(crow0 + r) * ldc + n0 + l15;
    if (mode == 0) {
      cp[0] = acc0[r]; cp[16] = acc1[r]; cp[32] = acc2[r]; cp[48] = acc3[r];
    } else {
      cp[0]  += acc0[r] + bs0; cp[16] += acc1[r] + bs1;
      cp[32] += acc2[r] + bs2; cp[48] += acc3[r] + bs3;
    }
  }
}

// ---------------------------------------------------------------------------
// Apply rotations to the 5 x H x 3 qkv vectors; add trans to Qd/Kd.
// ---------------------------------------------------------------------------
__global__ void rot_kernel(const float* __restrict__ qkv, const float* __restrict__ rotb,
                           const float* __restrict__ trb,
                           float* __restrict__ qr, float* __restrict__ kr,
                           float* __restrict__ qd, float* __restrict__ kd,
                           float* __restrict__ vv) {
  int id = blockIdx.x * 256 + threadIdx.x;
  if (id >= NTOK * H_) return;
  int t = id >> 4, h = id & 15;
  const float* R = rotb + (long)t * 9;
  const float* q = qkv + (long)t * 256;
  float out[5][3];
  for (int f = 0; f < 5; ++f) {
    float v0 = q[f * 48 + h * 3 + 0];
    float v1 = q[f * 48 + h * 3 + 1];
    float v2 = q[f * 48 + h * 3 + 2];
    for (int i = 0; i < 3; ++i)
      out[f][i] = R[i * 3 + 0] * v0 + R[i * 3 + 1] * v1 + R[i * 3 + 2] * v2;
  }
  const float* T = trb + (long)t * 3;
  long base = (long)id * 3;
  for (int i = 0; i < 3; ++i) {
    qr[base + i] = out[0][i];
    kr[base + i] = out[1][i];
    qd[base + i] = out[2][i] + T[i];
    kd[base + i] = out[3][i] + T[i];
    vv[base + i] = out[4][i];
  }
}

// ---------------------------------------------------------------------------
// Attention: one block (64 lanes) per (token, head); online softmax over j.
// ---------------------------------------------------------------------------
__global__ __launch_bounds__(64)
void attn_kernel(const float* __restrict__ qr, const float* __restrict__ kr,
                 const float* __restrict__ qd, const float* __restrict__ kd,
                 const float* __restrict__ vv, const unsigned char* __restrict__ maskb,
                 const float* __restrict__ r_s, const float* __restrict__ d_s,
                 float* __restrict__ o) {
  const int id = blockIdx.x;           // t*16 + h
  const int t = id >> 4, h = id & 15;
  const int b = t / L_;
  const int j0 = b * L_;
  const float scale = 0.57735026919f;  // 3^{-1/2}
  const float sr = softplusf(r_s[h]) * scale;
  const float sd = softplusf(d_s[h]) * scale;
  const long qb = (long)id * 3;
  const float q0 = qr[qb], q1 = qr[qb + 1], q2 = qr[qb + 2];
  const float p0 = qd[qb], p1 = qd[qb + 1], p2 = qd[qb + 2];
  const float mi_bias = maskb[t] ? -1.0e9f : 0.0f;

  float m = -1.0e30f, s = 0.0f, o0 = 0.0f, o1 = 0.0f, o2 = 0.0f;
  for (int j = threadIdx.x; j < L_; j += 64) {
    int tj = j0 + j;
    long kb = ((long)tj * 16 + h) * 3;
    float ar = (q0 * kr[kb] + q1 * kr[kb + 1] + q2 * kr[kb + 2]) * sr;
    float d0 = p0 - kd[kb], d1 = p1 - kd[kb + 1], d2 = p2 - kd[kb + 2];
    float dist = sqrtf(d0 * d0 + d1 * d1 + d2 * d2);
    float bias = maskb[tj] ? -1.0e9f : mi_bias;
    float logit = ar - dist * sd + bias;
    float mn = fmaxf(m, logit);
    float c = __expf(m - mn);
    float e = __expf(logit - mn);
    s = s * c + e;
    o0 = o0 * c + e * vv[kb];
    o1 = o1 * c + e * vv[kb + 1];
    o2 = o2 * c + e * vv[kb + 2];
    m = mn;
  }
  __shared__ float sm[64], ss[64], s0[64], s1[64], s2[64];
  int tid = threadIdx.x;
  sm[tid] = m; ss[tid] = s; s0[tid] = o0; s1[tid] = o1; s2[tid] = o2;
  for (int off = 32; off > 0; off >>= 1) {
    __syncthreads();
    if (tid < off) {
      float m2 = sm[tid + off];
      float mn = fmaxf(sm[tid], m2);
      float c1 = __expf(sm[tid] - mn);
      float c2 = __expf(m2 - mn);
      ss[tid] = ss[tid] * c1 + ss[tid + off] * c2;
      s0[tid] = s0[tid] * c1 + s0[tid + off] * c2;
      s1[tid] = s1[tid] * c1 + s1[tid + off] * c2;
      s2[tid] = s2[tid] * c1 + s2[tid + off] * c2;
      sm[tid] = mn;
    }
  }
  __syncthreads();
  if (tid == 0) {
    float inv = 1.0f / ss[0];
    o[qb]     = s0[0] * inv;
    o[qb + 1] = s1[0] * inv;
    o[qb + 2] = s2[0] * inv;
  }
}

// ---------------------------------------------------------------------------
// Inverse rotation (R^T) + pack to bf16 (NPAD x 64, cols 48..63 stay zero)
// ---------------------------------------------------------------------------
__global__ void invrot_kernel(const float* __restrict__ o, const float* __restrict__ rotb,
                              uint16_t* __restrict__ obuf) {
  int id = blockIdx.x * 256 + threadIdx.x;
  if (id >= NTOK * H_) return;
  int t = id >> 4, h = id & 15;
  const float* R = rotb + (long)t * 9;
  long base = (long)id * 3;
  float o0 = o[base], o1 = o[base + 1], o2 = o[base + 2];
  for (int i = 0; i < 3; ++i) {
    float v = R[0 * 3 + i] * o0 + R[1 * 3 + i] * o1 + R[2 * 3 + i] * o2;
    obuf[(long)t * 64 + h * 3 + i] = f2bf(v);
  }
}

// ---------------------------------------------------------------------------
// SwiGLU activation: g = silu(h[:, :2048]) * h[:, 2048:]
// ---------------------------------------------------------------------------
__global__ void swiglu_kernel(const float* __restrict__ h, uint16_t* __restrict__ g) {
  long idx = (long)blockIdx.x * 256 + threadIdx.x;
  if (idx >= (long)NPAD * HID_) return;
  long row = idx / HID_;
  int c = (int)(idx % HID_);
  float a = h[row * (2 * HID_) + c];
  float v = h[row * (2 * HID_) + HID_ + c];
  float sig = 1.0f / (1.0f + __expf(-a));
  g[idx] = f2bf(a * sig * v);
}

// ---------------------------------------------------------------------------
// Output heads: out = x + (e @ Wv^T) * sigmoid(e @ Wg^T + gb)  (unless masked)
// ---------------------------------------------------------------------------
__global__ __launch_bounds__(256)
void head_kernel(const float* __restrict__ emb,
                 const float* __restrict__ vh_x, const float* __restrict__ vl_x,
                 const float* __restrict__ wv_vh, const float* __restrict__ wg_vh,
                 const float* __restrict__ gb_vh,
                 const float* __restrict__ wv_vl, const float* __restrict__ wg_vl,
                 const float* __restrict__ gb_vl,
                 const unsigned char* __restrict__ vh_mask,
                 const unsigned char* __restrict__ vl_mask,
                 float* __restrict__ out) {
  int gid = blockIdx.x;              // 0 .. B_*(VH_+VL_)-1
  int b = gid / (VH_ + VL_), r = gid % (VH_ + VL_);
  int t; const float* x; const float* Wv; const float* Wg; const float* gb;
  float* op; unsigned char m;
  if (r < VH_) {
    t = b * L_ + r;
    x  = vh_x + ((long)b * VH_ + r) * DIN_;
    Wv = wv_vh; Wg = wg_vh; gb = gb_vh;
    op = out + ((long)b * VH_ + r) * DIN_;
    m  = vh_mask[b * VH_ + r];
  } else {
    int p = r - VH_;
    t = b * L_ + VH_ + p;
    x  = vl_x + ((long)b * VL_ + p) * DIN_;
    Wv = wv_vl; Wg = wg_vl; gb = gb_vl;
    op = out + (long)B_ * VH_ * DIN_ + ((long)b * VL_ + p) * DIN_;
    m  = vl_mask[b * VL_ + p];
  }
  __shared__ float e[D_];
  e[threadIdx.x]       = emb[(long)t * D_ + threadIdx.x];
  e[threadIdx.x + 256] = emb[(long)t * D_ + threadIdx.x + 256];
  __syncthreads();
  int d = threadIdx.x;
  const float* wv = Wv + (long)d * D_;
  const float* wg = Wg + (long)d * D_;
  float av = 0.0f, ag = 0.0f;
  for (int k = 0; k < D_; ++k) { av += wv[k] * e[k]; ag += wg[k] * e[k]; }
  float gate = 1.0f / (1.0f + __expf(-(ag + gb[d])));
  float xo = x[d];
  op[d] = m ? xo : xo + av * gate;
}

// ---------------------------------------------------------------------------
// Launcher
// ---------------------------------------------------------------------------
extern "C" void kernel_launch(void* const* d_in, const int* in_sizes, int n_in,
                              void* d_out, int out_size, void* d_ws, size_t ws_size,
                              hipStream_t stream) {
  (void)in_sizes; (void)n_in; (void)out_size; (void)ws_size;
  const float* vh_x      = (const float*)d_in[0];
  const float* vl_x      = (const float*)d_in[1];
  const float* vh_rots   = (const float*)d_in[2];
  const float* vh_trans  = (const float*)d_in[3];
  const float* vl_rots   = (const float*)d_in[4];
  const float* vl_trans  = (const float*)d_in[5];
  const float* ep_feats  = (const float*)d_in[6];
  const float* ep_rots   = (const float*)d_in[7];
  const float* ep_trans  = (const float*)d_in[8];
  const unsigned char* vh_mask = (const unsigned char*)d_in[9];
  const unsigned char* vl_mask = (const unsigned char*)d_in[10];
  const unsigned char* ep_mask = (const unsigned char*)d_in[11];
  const float* vh_emb_w  = (const float*)d_in[12];
  const float* vl_emb_w  = (const float*)d_in[13];
  const float* tgt_emb_w = (const float*)d_in[14];
  const float* qkv_w     = (const float*)d_in[15];
  const float* out_w     = (const float*)d_in[16];
  const float* out_b     = (const float*)d_in[17];
  const float* ln_g      = (const float*)d_in[18];
  const float* ln_b      = (const float*)d_in[19];
  const float* r_scale   = (const float*)d_in[20];
  const float* d_scale   = (const float*)d_in[21];
  const float* ln2_g     = (const float*)d_in[22];
  const float* ln2_b     = (const float*)d_in[23];
  const float* w12       = (const float*)d_in[24];
  const float* w3        = (const float*)d_in[25];
  const float* out_vh_w  = (const float*)d_in[26];
  const float* out_vh_gw = (const float*)d_in[27];
  const float* out_vh_gb = (const float*)d_in[28];
  const float* out_vl_w  = (const float*)d_in[29];
  const float* out_vl_gw = (const float*)d_in[30];
  const float* out_vl_gb = (const float*)d_in[31];

  char* wsp = (char*)d_ws;
  size_t off = 0;
  auto alloc = [&](size_t bytes) -> void* {
    void* p = wsp + off;
    off = (off + bytes + 255) & ~(size_t)255;
    return p;
  };
  uint16_t* wq   = (uint16_t*)alloc((size_t)NL_ * 256 * 512 * 2);
  uint16_t* wo   = (uint16_t*)alloc((size_t)NL_ * 512 * 64 * 2);
  uint16_t* w12b = (uint16_t*)alloc((size_t)NL_ * 4096 * 512 * 2);
  uint16_t* w3b  = (uint16_t*)alloc((size_t)NL_ * 512 * 2048 * 2);
  float*    emb  = (float*)   alloc((size_t)NPAD * D_ * 4);
  uint16_t* xn   = (uint16_t*)alloc((size_t)NPAD * D_ * 2);
  float*    qkvb = (float*)   alloc((size_t)NPAD * 256 * 4);
  float*    qr   = (float*)   alloc((size_t)NPAD * H_ * 3 * 4);
  float*    kr   = (float*)   alloc((size_t)NPAD * H_ * 3 * 4);
  float*    qdb  = (float*)   alloc((size_t)NPAD * H_ * 3 * 4);
  float*    kdb  = (float*)   alloc((size_t)NPAD * H_ * 3 * 4);
  float*    vvb  = (float*)   alloc((size_t)NPAD * H_ * 3 * 4);
  float*    oatt = (float*)   alloc((size_t)NPAD * H_ * 3 * 4);
  uint16_t* obuf = (uint16_t*)alloc((size_t)NPAD * 64 * 2);
  float*    hbuf = (float*)   alloc((size_t)NPAD * 4096 * 4);
  uint16_t* gbuf = (uint16_t*)alloc((size_t)NPAD * HID_ * 2);
  float*    rotb = (float*)   alloc((size_t)NPAD * 9 * 4);
  float*    trb  = (float*)   alloc((size_t)NPAD * 3 * 4);
  unsigned char* maskb = (unsigned char*)alloc((size_t)NPAD);

  // --- weight prep (bf16 + padding) ---
  auto cvt = [&](const float* s, uint16_t* dst, int batch, int sr, int dr, int sc, int dc) {
    long total = (long)batch * dr * dc;
    cvt_pad_kernel<<<(int)((total + 255) / 256), 256, 0, stream>>>(s, dst, batch, sr, dr, sc, dc);
  };
  cvt(qkv_w, wq,   NL_, 240, 256, 512, 512);     // pad 240->256 rows
  cvt(out_w, wo,   NL_, 512, 512, 48, 64);       // pad K 48->64
  cvt(w12,   w12b, NL_, 4096, 4096, 512, 512);
  cvt(w3,    w3b,  NL_, 512, 512, 2048, 2048);

  meta_kernel<<<NPAD / 256, 256, 0, stream>>>(vh_rots, vh_trans, vl_rots, vl_trans,
                                              ep_rots, ep_trans, vh_mask, vl_mask, ep_mask,
                                              rotb, trb, maskb);

  zero_f32_kernel<<<(NPAD * D_) / 256, 256, 0, stream>>>(emb, (long)NPAD * D_);
  zero_b16_kernel<<<(NPAD * 64) / 256, 256, 0, stream>>>(obuf, (long)NPAD * 64);

  embed_kernel<<<NTOK, 256, 0, stream>>>(vh_x, vl_x, ep_feats, vh_emb_w, vl_emb_w, tgt_emb_w, emb);

  const int rotBlocks = (NTOK * H_ + 255) / 256;
  for (int i = 0; i < NL_; ++i) {
    // Attention block
    ln_kernel<<<NPAD, 256, 0, stream>>>(emb, ln_g + (long)i * D_, ln_b + (long)i * D_, xn);
    gemm_wmma_kernel<<<dim3(256 / 64, NPAD / 64), 128, 0, stream>>>(
        xn, D_, wq + (long)i * 256 * 512, 512, qkvb, 256, nullptr, 512, 0);
    rot_kernel<<<rotBlocks, 256, 0, stream>>>(qkvb, rotb, trb, qr, kr, qdb, kdb, vvb);
    attn_kernel<<<NTOK * H_, 64, 0, stream>>>(qr, kr, qdb, kdb, vvb, maskb,
                                              r_scale + (long)i * H_, d_scale + (long)i * H_, oatt);
    invrot_kernel<<<rotBlocks, 256, 0, stream>>>(oatt, rotb, obuf);
    gemm_wmma_kernel<<<dim3(512 / 64, NPAD / 64), 128, 0, stream>>>(
        obuf, 64, wo + (long)i * 512 * 64, 64, emb, D_, out_b + (long)i * D_, 64, 1);
    // SwiGLU block
    ln_kernel<<<NPAD, 256, 0, stream>>>(emb, ln2_g + (long)i * D_, ln2_b + (long)i * D_, xn);
    gemm_wmma_kernel<<<dim3(4096 / 64, NPAD / 64), 128, 0, stream>>>(
        xn, D_, w12b + (long)i * 4096 * 512, 512, hbuf, 4096, nullptr, 512, 0);
    swiglu_kernel<<<(int)(((long)NPAD * HID_ + 255) / 256), 256, 0, stream>>>(hbuf, gbuf);
    gemm_wmma_kernel<<<dim3(512 / 64, NPAD / 64), 128, 0, stream>>>(
        gbuf, HID_, w3b + (long)i * 512 * 2048, 2048, emb, D_, nullptr, 2048, 2);
  }

  head_kernel<<<B_ * (VH_ + VL_), 256, 0, stream>>>(emb, vh_x, vl_x,
                                                    out_vh_w, out_vh_gw, out_vh_gb,
                                                    out_vl_w, out_vl_gw, out_vl_gb,
                                                    vh_mask, vl_mask, (float*)d_out);
}
